// WambaBlock_52312701665776
// MI455X (gfx1250) — compile-verified, matrix operand order checked by reading, not run on hardware
//
#include <hip/hip_runtime.h>
#include <math.h>

// ---------------- problem constants ----------------
#define Bb 2
#define Ss 1024
#define Hh 1024
#define Ii 2048
#define Nn 16
#define Rr 64
#define KC 4
#define NHh 16
#define NKVv 4
#define HDd 64
#define IMm 2816
#define W_ATT 512
#define W_SSM 128
#define TOK (Bb*Ss)          // 2048 tokens

// ---------------- WMMA types ----------------
typedef __bf16 bf16_t;
typedef bf16_t v16bf __attribute__((ext_vector_type(16)));
typedef float  v8f  __attribute__((ext_vector_type(8)));

__device__ __forceinline__ unsigned f2bf_pack(float lo, float hi) {
  // round-to-nearest (ties away) bf16 pair packed via v_perm_b32: 3 VALU ops
  unsigned a = __float_as_uint(lo) + 0x8000u;
  unsigned b = __float_as_uint(hi) + 0x8000u;
  return __builtin_amdgcn_perm(b, a, 0x07060302u);  // {b[31:24],b[23:16],a[31:24],a[23:16]}
}

__device__ __forceinline__ float siluf(float x) {
  return x / (1.0f + __expf(-x));
}

// ---------------- generic bf16 WMMA GEMM: C[M,N] = A[M,K] @ W[N,K]^T (+ optional residual) ----
// Assumptions (hold for every launch below): M % 128 == 0, K % 32 == 0,
// A/W/C rows 16B-aligned (lda/ldb/ldc % 4 == 0). N may be ragged (guarded).
#define BM 128
#define BN 64
#define BKk 32
#define BKP 40   // padded LDS row stride (bf16 elems); 80B = 5 * 16B -> aligned b128 frag loads

__global__ __launch_bounds__(256)
void gemm_bf16_wmma(const float* __restrict__ A, const float* __restrict__ W,
                    float* __restrict__ C, const float* __restrict__ Rp,
                    int M, int N, int K, int lda, int ldb, int ldc) {
  __shared__ __align__(16) unsigned short sA[BM * BKP];
  __shared__ __align__(16) unsigned short sB[BN * BKP];
  const int tid  = threadIdx.x;
  const int lane = tid & 31;
  const int wv   = tid >> 5;
  const int bm = blockIdx.y * BM;
  const int bn = blockIdx.x * BN;

  const int tmA = (wv & 3) * 2;     // wave's two m-tiles
  const int tnB = (wv >> 2) * 2;    // wave's two n-tiles
  const int halfsel = lane >> 4;    // which K-half this lane holds
  const int fr      = lane & 15;    // row (A: M, B: N) within 16x16 tile

  v8f acc00 = {0.f,0.f,0.f,0.f,0.f,0.f,0.f,0.f};
  v8f acc01 = acc00, acc10 = acc00, acc11 = acc00;

  // staging thread map: 8 float4 per row of 32 floats
  const int srow = tid >> 3;          // 0..31
  const int scol = (tid & 7) * 4;     // 0,4,...,28
  const int Nm1 = N - 1;
  int gnB0 = bn + srow;      if (gnB0 > Nm1) gnB0 = Nm1;   // clamp ragged N
  int gnB1 = bn + srow + 32; if (gnB1 > Nm1) gnB1 = Nm1;

  const float* aptr  = A + (size_t)(bm + srow) * lda + scol;
  const float* bptr0 = W + (size_t)gnB0 * ldb + scol;
  const float* bptr1 = W + (size_t)gnB1 * ldb + scol;
  const size_t astep = (size_t)32 * lda;

  float4 ra0, ra1, ra2, ra3, rb0g, rb1g;

  auto load_tiles = [&]() {
    ra0  = *(const float4*)(aptr);
    ra1  = *(const float4*)(aptr + astep);
    ra2  = *(const float4*)(aptr + 2 * astep);
    ra3  = *(const float4*)(aptr + 3 * astep);
    rb0g = *(const float4*)(bptr0);
    rb1g = *(const float4*)(bptr1);
    aptr += BKk; bptr0 += BKk; bptr1 += BKk;
  };
  auto store_tiles = [&]() {
    uint2 pk;
    pk.x = f2bf_pack(ra0.x, ra0.y);  pk.y = f2bf_pack(ra0.z, ra0.w);
    *(uint2*)(&sA[(srow +  0) * BKP + scol]) = pk;
    pk.x = f2bf_pack(ra1.x, ra1.y);  pk.y = f2bf_pack(ra1.z, ra1.w);
    *(uint2*)(&sA[(srow + 32) * BKP + scol]) = pk;
    pk.x = f2bf_pack(ra2.x, ra2.y);  pk.y = f2bf_pack(ra2.z, ra2.w);
    *(uint2*)(&sA[(srow + 64) * BKP + scol]) = pk;
    pk.x = f2bf_pack(ra3.x, ra3.y);  pk.y = f2bf_pack(ra3.z, ra3.w);
    *(uint2*)(&sA[(srow + 96) * BKP + scol]) = pk;
    pk.x = f2bf_pack(rb0g.x, rb0g.y); pk.y = f2bf_pack(rb0g.z, rb0g.w);
    *(uint2*)(&sB[(srow +  0) * BKP + scol]) = pk;
    pk.x = f2bf_pack(rb1g.x, rb1g.y); pk.y = f2bf_pack(rb1g.z, rb1g.w);
    *(uint2*)(&sB[(srow + 32) * BKP + scol]) = pk;
  };

  // fragment LDS addresses (loop-invariant)
  const uint4* pa = (const uint4*)sA;
  const uint4* pb = (const uint4*)sB;
  const int ra_i0 = (tmA * 16 + fr) * 5 + halfsel;   // row stride = BKP*2/16 = 5 uint4
  const int ra_i1 = ra_i0 + 16 * 5;
  const int rb_i0 = (tnB * 16 + fr) * 5 + halfsel;
  const int rb_i1 = rb_i0 + 16 * 5;

  // software pipeline: tile k in LDS, tile k+1 in registers
  load_tiles();
  store_tiles();

  for (int k0 = 0;;) {
    __syncthreads();                 // LDS tile ready for all waves
    const int knext = k0 + BKk;
    const bool more = knext < K;
    if (more) load_tiles();          // global loads for next tile, overlap WMMA

    union { v16bf v; uint4 u[2]; } fa0, fa1, fb0, fb1;
    fa0.u[0] = pa[ra_i0];  fa0.u[1] = pa[ra_i0 + 2];
    fa1.u[0] = pa[ra_i1];  fa1.u[1] = pa[ra_i1 + 2];
    fb0.u[0] = pb[rb_i0];  fb0.u[1] = pb[rb_i0 + 2];
    fb1.u[0] = pb[rb_i1];  fb1.u[1] = pb[rb_i1 + 2];

    acc00 = __builtin_amdgcn_wmma_f32_16x16x32_bf16(false, fa0.v, false, fb0.v,
                                                    (short)0, acc00, false, false);
    acc01 = __builtin_amdgcn_wmma_f32_16x16x32_bf16(false, fa0.v, false, fb1.v,
                                                    (short)0, acc01, false, false);
    acc10 = __builtin_amdgcn_wmma_f32_16x16x32_bf16(false, fa1.v, false, fb0.v,
                                                    (short)0, acc10, false, false);
    acc11 = __builtin_amdgcn_wmma_f32_16x16x32_bf16(false, fa1.v, false, fb1.v,
                                                    (short)0, acc11, false, false);
    if (!more) break;
    __syncthreads();                 // all waves done reading this LDS tile
    store_tiles();                   // commit next tile to LDS
    k0 = knext;
  }

  // D layout: VGPR j -> M = j + 8*halfsel ; N = lane&15
  const int col0 = bn + tnB * 16 + fr;
  const int col1 = col0 + 16;
  const int row0 = bm + tmA * 16 + halfsel * 8;
  const int row1 = row0 + 16;
  const bool c0ok = (col0 < N), c1ok = (col1 < N);
  if (Rp) {          // fused residual add (uniform branch)
#pragma unroll
    for (int j = 0; j < 8; ++j) {
      size_t i00 = (size_t)(row0 + j) * ldc + col0;
      size_t i01 = (size_t)(row0 + j) * ldc + col1;
      size_t i10 = (size_t)(row1 + j) * ldc + col0;
      size_t i11 = (size_t)(row1 + j) * ldc + col1;
      if (c0ok) C[i00] = acc00[j] + Rp[i00];
      if (c1ok) C[i01] = acc01[j] + Rp[i01];
      if (c0ok) C[i10] = acc10[j] + Rp[i10];
      if (c1ok) C[i11] = acc11[j] + Rp[i11];
    }
  } else {
#pragma unroll
    for (int j = 0; j < 8; ++j) {
      if (c0ok) C[(size_t)(row0 + j) * ldc + col0] = acc00[j];
      if (c1ok) C[(size_t)(row0 + j) * ldc + col1] = acc01[j];
      if (c0ok) C[(size_t)(row1 + j) * ldc + col0] = acc10[j];
      if (c1ok) C[(size_t)(row1 + j) * ldc + col1] = acc11[j];
    }
  }
}

// ---------------- RMSNorm: per-row (H-wide) ----------------
__global__ __launch_bounds__(256)
void rmsnorm_kernel(const float* __restrict__ x, const float* __restrict__ w,
                    float* __restrict__ out, int Hdim) {
  __shared__ float red[256];
  const int row = blockIdx.x, tid = threadIdx.x;
  const float* xr = x + (size_t)row * Hdim;
  float s = 0.f;
  for (int i = tid; i < Hdim; i += 256) { float v = xr[i]; s += v * v; }
  red[tid] = s; __syncthreads();
  for (int k = 128; k > 0; k >>= 1) { if (tid < k) red[tid] += red[tid + k]; __syncthreads(); }
  float r = rsqrtf(red[0] / (float)Hdim + 1e-6f);
  float* orow = out + (size_t)row * Hdim;
  for (int i = tid; i < Hdim; i += 256) orow[i] = xr[i] * r * w[i];
}

// ---------------- causal conv (K=4) + SiLU ----------------
__global__ __launch_bounds__(256)
void conv_silu_kernel(const float* __restrict__ xz, const float* __restrict__ cw,
                      const float* __restrict__ cb, float* __restrict__ xs) {
  int idx = blockIdx.x * 256 + threadIdx.x;
  if (idx >= Bb * Ss * Ii) return;
  int i = idx % Ii, t = (idx / Ii) % Ss, b = idx / (Ii * Ss);
  float acc = cb[i];
#pragma unroll
  for (int k = 0; k < KC; ++k) {
    int tt = t + k - (KC - 1);
    if (tt >= 0) acc += cw[i * KC + k] * xz[(size_t)(b * Ss + tt) * (2 * Ii) + i];
  }
  xs[idx] = siluf(acc);
}

// ---------------- dt bias + softplus ----------------
__global__ __launch_bounds__(256)
void softplus_bias_kernel(float* __restrict__ dt, const float* __restrict__ bias) {
  int idx = blockIdx.x * 256 + threadIdx.x;
  if (idx >= TOK * Ii) return;
  float v = dt[idx] + bias[idx % Ii];
  dt[idx] = (v > 20.0f) ? v : log1pf(__expf(v));
}

// ---------------- selective scan: one thread per (b, channel) ----------------
__global__ __launch_bounds__(256)
void scan_kernel(const float* __restrict__ dt, const float* __restrict__ xs,
                 const float* __restrict__ proj, const float* __restrict__ A_log,
                 float* __restrict__ y) {
  int idx = blockIdx.x * 256 + threadIdx.x;
  if (idx >= Bb * Ii) return;
  int b = idx / Ii, i = idx % Ii;
  float Ai[Nn], h[Nn];
#pragma unroll
  for (int n = 0; n < Nn; ++n) { Ai[n] = -__expf(A_log[i * Nn + n]); h[n] = 0.f; }
  for (int t = 0; t < Ss; ++t) {
    size_t r = (size_t)(b * Ss + t);
    float d  = dt[r * Ii + i];
    float xv = xs[r * Ii + i];
    const float* pr = proj + r * (Rr + 2 * Nn);
    const float* Bm = pr + Rr;
    const float* Cm = pr + Rr + Nn;
    float acc = 0.f;
#pragma unroll
    for (int n = 0; n < Nn; ++n) {
      h[n] = h[n] * __expf(d * Ai[n]) + d * Bm[n] * xv;
      acc += h[n] * Cm[n];
    }
    y[r * Ii + i] = acc;
  }
}

// ---------------- y = (y + D*xs) * silu(z) ----------------
__global__ __launch_bounds__(256)
void ygate_kernel(float* __restrict__ y, const float* __restrict__ xs,
                  const float* __restrict__ xz, const float* __restrict__ Dp) {
  int idx = blockIdx.x * 256 + threadIdx.x;
  if (idx >= TOK * Ii) return;
  int i = idx % Ii; size_t row = idx / Ii;
  float z = xz[row * (2 * Ii) + Ii + i];
  y[idx] = (y[idx] + Dp[i] * xs[idx]) * siluf(z);
}

// ---------------- rotary (in place), pair (j, j+32) per thread ----------------
__global__ __launch_bounds__(256)
void rotary_kernel(float* __restrict__ x, int nheads) {
  int idx = blockIdx.x * 256 + threadIdx.x;
  int total = Bb * Ss * nheads * (HDd / 2);
  if (idx >= total) return;
  int j = idx & 31;
  int h = (idx >> 5) % nheads;
  int s = (idx / (32 * nheads)) % Ss;
  int b = idx / (32 * nheads * Ss);
  float inv = __powf(10000.0f, -(2.0f * (float)j) / (float)HDd);
  float f = (float)s * inv;
  float c = __cosf(f), sn = __sinf(f);
  float* p = x + ((size_t)(b * Ss + s) * nheads + h) * HDd;
  float x1 = p[j], x2 = p[j + 32];
  p[j]      = x1 * c - x2 * sn;
  p[j + 32] = x2 * c + x1 * sn;
}

// ---------------- windowed dual-KV attention with joint softmax ----------------
__global__ __launch_bounds__(64)
void attn_kernel(const float* __restrict__ q,
                 const float* __restrict__ ka, const float* __restrict__ va,
                 const float* __restrict__ ks, const float* __restrict__ vs,
                 float* __restrict__ o) {
  const int qi = blockIdx.x, h = blockIdx.y, b = blockIdx.z;
  const int kvh = h >> 2;   // NH/NKV = 4
  __shared__ float sq[HDd];
  __shared__ float sc[W_ATT + W_SSM];
  __shared__ float red[64];
  const int tid = threadIdx.x;
  const float scale = 0.125f;   // HD^-0.5

  sq[tid] = q[((size_t)(b * Ss + qi) * NHh + h) * HDd + tid];
  __syncthreads();

  int a0 = qi - (W_ATT - 1); if (a0 < 0) a0 = 0; int na = qi - a0 + 1;
  int s0 = qi - (W_SSM - 1); if (s0 < 0) s0 = 0; int ns = qi - s0 + 1;

  for (int i = tid; i < na; i += 64) {
    const float* kp = ka + ((size_t)(b * Ss + a0 + i) * NKVv + kvh) * HDd;
    float acc = 0.f;
#pragma unroll
    for (int d = 0; d < HDd; ++d) acc += sq[d] * kp[d];
    sc[i] = acc * scale;
  }
  for (int i = tid; i < ns; i += 64) {
    const float* kp = ks + ((size_t)(b * Ss + s0 + i) * NKVv + kvh) * HDd;
    float acc = 0.f;
#pragma unroll
    for (int d = 0; d < HDd; ++d) acc += sq[d] * kp[d];
    sc[W_ATT + i] = acc * scale;
  }
  for (int i = tid; i < W_ATT + W_SSM; i += 64) {
    if ((i < W_ATT && i >= na) || (i >= W_ATT && (i - W_ATT) >= ns)) sc[i] = -1e30f;
  }
  __syncthreads();

  float mx = -1e30f;
  for (int i = tid; i < W_ATT + W_SSM; i += 64) mx = fmaxf(mx, sc[i]);
  red[tid] = mx; __syncthreads();
  for (int k = 32; k > 0; k >>= 1) { if (tid < k) red[tid] = fmaxf(red[tid], red[tid + k]); __syncthreads(); }
  float mxv = red[0]; __syncthreads();

  float sum = 0.f;
  for (int i = tid; i < W_ATT + W_SSM; i += 64) { float e = __expf(sc[i] - mxv); sc[i] = e; sum += e; }
  red[tid] = sum; __syncthreads();
  for (int k = 32; k > 0; k >>= 1) { if (tid < k) red[tid] += red[tid + k]; __syncthreads(); }
  float invd = 1.0f / red[0];

  // phase 2: one output dim per thread
  float acc = 0.f;
  for (int i = 0; i < na; ++i)
    acc += sc[i] * va[((size_t)(b * Ss + a0 + i) * NKVv + kvh) * HDd + tid];
  for (int i = 0; i < ns; ++i)
    acc += sc[W_ATT + i] * vs[((size_t)(b * Ss + s0 + i) * NKVv + kvh) * HDd + tid];
  o[((size_t)(b * Ss + qi) * NHh + h) * HDd + tid] = acc * invd;
}

// ---------------- SwiGLU: h1 = silu(g[:, :IM]) * g[:, IM:] ----------------
__global__ __launch_bounds__(256)
void swiglu_kernel(const float* __restrict__ g, float* __restrict__ h1) {
  int idx = blockIdx.x * 256 + threadIdx.x;
  if (idx >= TOK * IMm) return;
  size_t r = idx / IMm; int j = idx % IMm;
  float g1 = g[r * (2 * IMm) + j];
  float g2 = g[r * (2 * IMm) + IMm + j];
  h1[idx] = siluf(g1) * g2;
}

// ---------------- host-side orchestration ----------------
static inline dim3 gemm_grid(int Mr, int Nc) { return dim3((Nc + BN - 1) / BN, (Mr + BM - 1) / BM); }
#define EW_GRID(n) dim3(((n) + 255) / 256)

extern "C" void kernel_launch(void* const* d_in, const int* in_sizes, int n_in,
                              void* d_out, int out_size, void* d_ws, size_t ws_size,
                              hipStream_t stream) {
  (void)in_sizes; (void)n_in; (void)out_size; (void)ws_size;
  const float* hidden   = (const float*)d_in[0];
  const float* pre_w    = (const float*)d_in[1];
  const float* in_proj  = (const float*)d_in[2];
  const float* conv_w   = (const float*)d_in[3];
  const float* conv_b   = (const float*)d_in[4];
  const float* x_proj   = (const float*)d_in[5];
  const float* dt_w     = (const float*)d_in[6];
  const float* dt_b     = (const float*)d_in[7];
  const float* A_log    = (const float*)d_in[8];
  const float* Dp       = (const float*)d_in[9];
  const float* out_proj = (const float*)d_in[10];
  const float* ssm_nw   = (const float*)d_in[11];
  const float* q_w      = (const float*)d_in[12];
  const float* k_w      = (const float*)d_in[13];
  const float* v_w      = (const float*)d_in[14];
  const float* o_w      = (const float*)d_in[15];
  const float* mlp_nw   = (const float*)d_in[16];
  const float* gate_w   = (const float*)d_in[17];
  const float* down_w   = (const float*)d_in[18];
  float* out = (float*)d_out;

  // workspace layout (offsets in floats)
  const size_t MF = (size_t)1 << 20;
  float* ws = (float*)d_ws;
  float* hs      = ws + 0;                 // 2M
  float* xz      = ws + 2 * MF;            // 8M
  float* xs      = ws + 10 * MF;           // 4M
  float* proj    = ws + 14 * MF;           // 0.25M
  float* dtb     = ws + 14 * MF + MF / 4;  // 4M
  float* yb      = dtb + 4 * MF;           // 4M
  float* ssm_res = yb + 4 * MF;            // 2M
  float* ssm_st  = ssm_res + 2 * MF;       // 2M
  float* qb      = ssm_st + 2 * MF;        // 2M
  float* ka      = qb + 2 * MF;            // 0.5M
  float* va      = ka + MF / 2;            // 0.5M
  float* ksb     = va + MF / 2;            // 0.5M
  float* vsb     = ksb + MF / 2;           // 0.5M
  float* ob      = vsb + MF / 2;           // 2M
  float* mlp_res = ob + 2 * MF;            // 2M
  float* x2      = mlp_res + 2 * MF;       // 2M
  float* gbuf    = xz;                     // reuse xz+xs region (needs 11M, have 12M)
  float* h1      = dtb;                    // reuse dt+y region (needs 5.5M, have 8M)

  // 1. pre-norm
  rmsnorm_kernel<<<TOK, 256, 0, stream>>>(hidden, pre_w, hs, Hh);
  // 2. in_proj: (2048 x 4096) = hs @ in_proj^T
  gemm_bf16_wmma<<<gemm_grid(TOK, 2 * Ii), 256, 0, stream>>>(hs, in_proj, xz, nullptr, TOK, 2 * Ii, Hh, Hh, Hh, 2 * Ii);
  // 3. conv + silu
  conv_silu_kernel<<<EW_GRID(TOK * Ii), 256, 0, stream>>>(xz, conv_w, conv_b, xs);
  // 4. x_proj: (2048 x 96)
  gemm_bf16_wmma<<<gemm_grid(TOK, Rr + 2 * Nn), 256, 0, stream>>>(xs, x_proj, proj, nullptr, TOK, Rr + 2 * Nn, Ii, Ii, Ii, Rr + 2 * Nn);
  // 5. dt_proj: (2048 x 2048), A = proj[:, :R] (lda = 96)
  gemm_bf16_wmma<<<gemm_grid(TOK, Ii), 256, 0, stream>>>(proj, dt_w, dtb, nullptr, TOK, Ii, Rr, Rr + 2 * Nn, Rr, Ii);
  // 6. softplus(dt + bias)
  softplus_bias_kernel<<<EW_GRID(TOK * Ii), 256, 0, stream>>>(dtb, dt_b);
  // 7. selective scan
  scan_kernel<<<EW_GRID(Bb * Ii), 256, 0, stream>>>(dtb, xs, proj, A_log, yb);
  // 8. y = (y + D*xs) * silu(z)
  ygate_kernel<<<EW_GRID(TOK * Ii), 256, 0, stream>>>(yb, xs, xz, Dp);
  // 9+10. out_proj with fused residual: ssm_res = yb @ out_proj^T + hidden
  gemm_bf16_wmma<<<gemm_grid(TOK, Hh), 256, 0, stream>>>(yb, out_proj, ssm_res, hidden, TOK, Hh, Ii, Ii, Ii, Hh);
  // 11. ssm norm
  rmsnorm_kernel<<<TOK, 256, 0, stream>>>(ssm_res, ssm_nw, ssm_st, Hh);
  // 12-16. attention projections
  gemm_bf16_wmma<<<gemm_grid(TOK, NHh * HDd), 256, 0, stream>>>(hs, q_w, qb, nullptr, TOK, NHh * HDd, Hh, Hh, Hh, NHh * HDd);
  gemm_bf16_wmma<<<gemm_grid(TOK, NKVv * HDd), 256, 0, stream>>>(hs, k_w, ka, nullptr, TOK, NKVv * HDd, Hh, Hh, Hh, NKVv * HDd);
  gemm_bf16_wmma<<<gemm_grid(TOK, NKVv * HDd), 256, 0, stream>>>(hs, v_w, va, nullptr, TOK, NKVv * HDd, Hh, Hh, Hh, NKVv * HDd);
  gemm_bf16_wmma<<<gemm_grid(TOK, NKVv * HDd), 256, 0, stream>>>(ssm_st, k_w, ksb, nullptr, TOK, NKVv * HDd, Hh, Hh, Hh, NKVv * HDd);
  gemm_bf16_wmma<<<gemm_grid(TOK, NKVv * HDd), 256, 0, stream>>>(ssm_st, v_w, vsb, nullptr, TOK, NKVv * HDd, Hh, Hh, Hh, NKVv * HDd);
  // 17. rotary (q, k_att, k_ssm)
  rotary_kernel<<<EW_GRID(Bb * Ss * NHh * 32), 256, 0, stream>>>(qb, NHh);
  rotary_kernel<<<EW_GRID(Bb * Ss * NKVv * 32), 256, 0, stream>>>(ka, NKVv);
  rotary_kernel<<<EW_GRID(Bb * Ss * NKVv * 32), 256, 0, stream>>>(ksb, NKVv);
  // 18. windowed attention (joint softmax over att+ssm paths)
  attn_kernel<<<dim3(Ss, NHh, Bb), 64, 0, stream>>>(qb, ka, va, ksb, vsb, ob);
  // 19+20. o projection with fused residual: mlp_res = ob @ o_w^T + ssm_res
  gemm_bf16_wmma<<<gemm_grid(TOK, Hh), 256, 0, stream>>>(ob, o_w, mlp_res, ssm_res, TOK, Hh, NHh * HDd, NHh * HDd, NHh * HDd, Hh);
  // 21. mlp norm
  rmsnorm_kernel<<<TOK, 256, 0, stream>>>(mlp_res, mlp_nw, x2, Hh);
  // 22. gate/up projection (2048 x 5632)
  gemm_bf16_wmma<<<gemm_grid(TOK, 2 * IMm), 256, 0, stream>>>(x2, gate_w, gbuf, nullptr, TOK, 2 * IMm, Hh, Hh, Hh, 2 * IMm);
  // 23. swiglu
  swiglu_kernel<<<EW_GRID(TOK * IMm), 256, 0, stream>>>(gbuf, h1);
  // 24+25. down projection with fused residual -> final output
  gemm_bf16_wmma<<<gemm_grid(TOK, Hh), 256, 0, stream>>>(h1, down_w, out, mlp_res, TOK, Hh, IMm, IMm, IMm, Hh);
}
